// ConvColumn_38156489458340
// MI455X (gfx1250) — matrix operand order repaired and checked by reading, not compiled, not run on hardware
//
#include <hip/hip_runtime.h>
#include <hip/hip_bf16.h>

// ---------------------------------------------------------------------------
// Problem constants (from the reference)
// ---------------------------------------------------------------------------
#define BB     4      // batch
#define CIN    2
#define HH     48
#define WW     48
#define TT     96     // input time
#define COUT   64
#define KH     3
#define KW     3
#define KS     48     // temporal kernel length (= padding = fodep)
#define XO     23     // (48-3)/2+1
#define YO     23
#define NS     (XO*YO)          // 529 sites
#define TO     145              // output time = 96 + 2*48 - 48 + 1
#define KTOT   (CIN*KH*KW*KS)   // 864 reduction length
#define KTILES (KTOT/32)        // 27 WMMA K-steps
#define TCHUNK 16               // time tile (= WMMA N)
#define THETA  5.4f
#define WINCAP 265              // ceil(0.5*529)
#define FODEP  48
#define NSEG   (CIN*KH*KW)      // 18 input lines per site
#define STW    64               // staged window: t' in [toff-48, toff+15]
#define SELEM  (NSEG*STW)       // 1152 staged halves per site
#define SPB    265              // site pairs per batch (529 -> 265, last padded)

// ws layout (bytes)
#define AW_OFF   0              // 4*27*32*16 halves = 110592 B
#define DEP_OFF  110592         // 4*529 ints       = 8464 B
#define POT_OFF  131072         // 4*16*529*64 f32  = 8667136 B  (total ~8.8 MB)

typedef __attribute__((ext_vector_type(16))) _Float16 v16h;
typedef __attribute__((ext_vector_type(8)))  float    v8f;

// ---------------------------------------------------------------------------
// Build A = step-fire-leak(weight), time-flipped, f16, pre-swizzled into the
// WMMA A-fragment layout, contiguous k order matched by B (16x32 per tile):
//   lane<16 : M=lane,   K = h
//   lane>=16: M=lane-16,K = 16+h
// Stored as Aw[((mt*27+kt)*32+lane)*16 + h] so the GEMM loads one v16h/lane.
// ---------------------------------------------------------------------------
__global__ void build_a_kernel(const float* __restrict__ weight,
                               _Float16* __restrict__ Aw) {
    int u = blockIdx.x * blockDim.x + threadIdx.x;
    if (u >= 4 * KTILES * 32) return;
    int lane = u & 31;
    int kt   = (u >> 5) % KTILES;
    int mt   = u / (KTILES * 32);
    int lanelo = lane & 15, lanehi = lane >> 4;
    int o = mt * 16 + lanelo;                 // output channel (M row)
    #pragma unroll
    for (int h = 0; h < 16; ++h) {
        int k = kt * 32 + lanehi * 16 + h;
        int seg = k / KS, ktt = k % KS;       // tap index within 48
        int ci = seg / 9, rr = seg % 9, ky = rr / 3, kx = rr % 3;
        float w  = weight[((o * CIN + ci) * KH + ky) * KW + kx];
        float tk = (float)(KS - 1 - ktt);     // flipped time
        // max(0, min(t/16, -(t - 16w)/32 + w))
        float v = fminf(tk * 0.0625f, (w * 16.0f - tk) * 0.03125f + w);
        v = fmaxf(v, 0.0f);
        Aw[u * 16 + h] = (_Float16)v;
    }
}

// ---------------------------------------------------------------------------
// Zero the one-hot output and the depression state (fresh every launch).
// ---------------------------------------------------------------------------
__global__ void zero_kernel(float* __restrict__ out, int n4,
                            int* __restrict__ dep, int ndep) {
    int gid = blockIdx.x * blockDim.x + threadIdx.x;
    int stride = gridDim.x * blockDim.x;
    float4* o4 = (float4*)out;
    float4 z = make_float4(0.f, 0.f, 0.f, 0.f);
    for (int i = gid; i < n4; i += stride) o4[i] = z;
    if (gid < ndep) dep[gid] = 0;
}

// ---------------------------------------------------------------------------
// Implicit-GEMM conv via WMMA. One wave = TWO sites x (64ch x 16 t') tiles.
// Per site, the 18 input lines x 64 time samples covering every tap of this
// time chunk are staged once into LDS as f16 (pad-masked); inner loop builds
// B fragments with plain 16-bit DS loads (a 16-aligned K-run never crosses a
// 48-tap segment since 48 % 16 == 0). Each A fragment feeds 2 WMMAs.
// ---------------------------------------------------------------------------
__global__ __launch_bounds__(256)
void conv_wmma_kernel(const float* __restrict__ xin,
                      const float* __restrict__ bias,
                      const _Float16* __restrict__ Aw,
                      float* __restrict__ pot, int toff) {
    __shared__ _Float16 stage[8][2][SELEM];   // 36 KB per block
    int gid  = blockIdx.x * blockDim.x + threadIdx.x;
    int wave = gid >> 5;
    int lane = threadIdx.x & 31;
    int wl   = threadIdx.x >> 5;
    if (wave >= BB * SPB) return;             // wave-uniform exit

    int pr = wave % SPB;
    int b  = wave / SPB;
    int s0 = 2 * pr;
    int s1 = 2 * pr + 1;
    bool s1ok = (s1 < NS);                    // wave-uniform
    int s1c = s1ok ? s1 : s0;                 // duplicate compute when padded
    int lanelo = lane & 15;
    int lanehi = lane >> 4;
    int tprime = toff + lanelo;               // N column = output timestep

    // ---- stage both site windows: f32 -> f16, time-pad masked, once ----
    #pragma unroll
    for (int pair = 0; pair < 2; ++pair) {
        int s = pair ? s1c : s0;
        int sx = s / YO, sy = s % YO;
        for (int e = lane; e < SELEM; e += 32) {   // 1152 % 32 == 0: uniform
            int L = e >> 6, tloc = e & 63;
            int ci = L / 9, rr = L % 9, ky = rr / 3, kx = rr % 3;
            int t = toff - KS + tloc;
            float v = (t >= 0 && t < TT)
                ? xin[(((size_t)(b * CIN + ci) * HH + (2 * sx + ky)) * WW
                       + (2 * sy + kx)) * TT + t]
                : 0.0f;
            stage[wl][pair][e] = (_Float16)v;
        }
    }

    v8f acc[2][4] = {};
    const v16h* Av = (const v16h*)Aw;
    const _Float16* st0 = stage[wl][0];
    const _Float16* st1 = stage[wl][1];

    for (int kt = 0; kt < KTILES; ++kt) {
        int klo = kt * 32 + lanehi * 16;      // klo % 48 in {0,16,32}
        int seg = klo / KS;
        int kt0 = klo % KS;
        int hb  = seg * STW + lanelo + kt0;   // base staged-half index
        v16h b0, b1;
        #pragma unroll
        for (int j = 0; j < 16; ++j) {
            b0[j] = st0[hb + j];
            b1[j] = st1[hb + j];
        }
        #pragma unroll
        for (int mt = 0; mt < 4; ++mt) {
            v16h a = Av[(mt * KTILES + kt) * 32 + lane];
            acc[0][mt] = __builtin_amdgcn_wmma_f32_16x16x32_f16(
                false, a, false, b0, (short)0, acc[0][mt], false, false);
            acc[1][mt] = __builtin_amdgcn_wmma_f32_16x16x32_f16(
                false, a, false, b1, (short)0, acc[1][mt], false, false);
        }
    }

    if (tprime < TO) {
        #pragma unroll
        for (int pair = 0; pair < 2; ++pair) {
            if (pair == 1 && !s1ok) break;    // wave-uniform
            int s = pair ? s1 : s0;
            size_t base = ((size_t)((b * TCHUNK + lanelo) * NS + s)) * COUT
                          + lanehi * 8;       // 32B aligned
            #pragma unroll
            for (int mt = 0; mt < 4; ++mt) {
                int c = mt * 16 + lanehi * 8; // C/D layout: M = r + 8*hi
                float4 lo = make_float4(acc[pair][mt][0] + bias[c + 0],
                                        acc[pair][mt][1] + bias[c + 1],
                                        acc[pair][mt][2] + bias[c + 2],
                                        acc[pair][mt][3] + bias[c + 3]);
                float4 hi = make_float4(acc[pair][mt][4] + bias[c + 4],
                                        acc[pair][mt][5] + bias[c + 5],
                                        acc[pair][mt][6] + bias[c + 6],
                                        acc[pair][mt][7] + bias[c + 7]);
                float4* q = (float4*)(pot + base + mt * 16);
                q[0] = lo;
                q[1] = hi;
            }
        }
    }
}

// ---------------------------------------------------------------------------
// Winner-takes-all over one time chunk. One block per batch; thread = site.
// Per-batch depressed-site count via LDS atomic + barriers each timestep.
// Depression state persists across chunks in ws (zeroed each launch).
// ---------------------------------------------------------------------------
__global__ __launch_bounds__(544)
void wta_kernel(const float* __restrict__ pot, float* __restrict__ out,
                int* __restrict__ dep, int toff) {
    int b   = blockIdx.x;
    int tid = threadIdx.x;
    bool isSite = tid < NS;
    int s = tid;
    __shared__ int cnt;
    int myDep = isSite ? dep[b * NS + s] : 0;
    int tend = toff + TCHUNK; if (tend > TO) tend = TO;

    for (int t = toff; t < tend; ++t) {
        if (tid == 0) cnt = 0;
        __syncthreads();
        if (isSite && myDep != 0) atomicAdd(&cnt, 1);
        __syncthreads();
        bool k_ok = (cnt < WINCAP);
        int spike = 0;
        if (isSite && myDep == 0 && k_ok) {
            const float4* p4 = (const float4*)
                (pot + ((size_t)((b * TCHUNK + (t - toff)) * NS + s)) * COUT);
            float best = -1.0f; int wi = 0;   // potentials >= bias > 0
            #pragma unroll
            for (int cc = 0; cc < 16; ++cc) { // strict '>' keeps first-max
                float4 v = p4[cc];
                if (v.x > best) { best = v.x; wi = 4 * cc + 0; }
                if (v.y > best) { best = v.y; wi = 4 * cc + 1; }
                if (v.z > best) { best = v.z; wi = 4 * cc + 2; }
                if (v.w > best) { best = v.w; wi = 4 * cc + 3; }
            }
            if (best > THETA) {
                spike = 1;
                out[((size_t)(b * COUT + wi) * NS + s) * TO + t] = 1.0f;
            }
        }
        myDep += spike * FODEP - 1;
        myDep = myDep < 0 ? 0 : (myDep > FODEP - 1 ? FODEP - 1 : myDep);
        __syncthreads();   // protect cnt reset for next iteration
    }
    if (isSite) dep[b * NS + s] = myDep;
}

// ---------------------------------------------------------------------------
extern "C" void kernel_launch(void* const* d_in, const int* in_sizes, int n_in,
                              void* d_out, int out_size, void* d_ws, size_t ws_size,
                              hipStream_t stream) {
    const float* x      = (const float*)d_in[0];   // [4,2,48,48,96]
    const float* weight = (const float*)d_in[1];   // [64,2,3,3]
    const float* bias   = (const float*)d_in[2];   // [64]
    float* out = (float*)d_out;                    // [4,64,23,23,145]

    _Float16* Aw  = (_Float16*)((char*)d_ws + AW_OFF);
    int*      dep = (int*)((char*)d_ws + DEP_OFF);
    float*    pot = (float*)((char*)d_ws + POT_OFF);

    // 1) weights -> WMMA-layout f16 A matrix (108 KB)
    build_a_kernel<<<(4 * KTILES * 32 + 255) / 256, 256, 0, stream>>>(weight, Aw);
    // 2) zero output + depression state
    zero_kernel<<<2048, 256, 0, stream>>>(out, out_size / 4, dep, BB * NS);
    // 3) 10 time chunks: WMMA conv chunk -> sequential WTA chunk
    int conv_blocks = (BB * SPB * 32 + 255) / 256;  // 1060 waves
    for (int toff = 0; toff < TO; toff += TCHUNK) {
        conv_wmma_kernel<<<conv_blocks, 256, 0, stream>>>(x, bias, Aw, pot, toff);
        wta_kernel<<<BB, 544, 0, stream>>>(pot, out, dep, toff);
    }
}
